// TGNN_41369124995826
// MI455X (gfx1250) — matrix operand order repaired and checked by reading, not compiled, not run on hardware
//
#include <hip/hip_runtime.h>
#include <stdint.h>

// ---------------- WMMA types & helpers (gfx1250, wave32) ----------------
typedef __attribute__((ext_vector_type(16))) __bf16 v16bf;
typedef __attribute__((ext_vector_type(8)))  float  v8f;

union Frag16 { v16bf v; unsigned int u[8]; };

// ---- float -> bf16 conversion: prefer HW packed cvt, fallback to truncation ----
#if __has_builtin(__builtin_amdgcn_cvt_pk_bf16_f32)
typedef __attribute__((ext_vector_type(2))) __bf16 v2bf;
__device__ __forceinline__ unsigned int pack2(float lo, float hi) {
    return __builtin_bit_cast(unsigned int, __builtin_amdgcn_cvt_pk_bf16_f32(lo, hi));
}
__device__ __forceinline__ unsigned short f2bf(float f) {
    return (unsigned short)__builtin_bit_cast(unsigned int,
                                              __builtin_amdgcn_cvt_pk_bf16_f32(f, 0.0f));
}
#else
__device__ __forceinline__ unsigned int pack2(float lo, float hi) {
    unsigned int ul = __builtin_bit_cast(unsigned int, lo);
    unsigned int uh = __builtin_bit_cast(unsigned int, hi);
    return (ul >> 16) | (uh & 0xffff0000u);   // truncate to bf16, 2 VALU ops
}
__device__ __forceinline__ unsigned short f2bf(float f) {
    return (unsigned short)(__builtin_bit_cast(unsigned int, f) >> 16);
}
#endif

__device__ __forceinline__ float bf2f(unsigned short b) {
    union { unsigned int u; float f; } c; c.u = ((unsigned int)b) << 16;
    return c.f;
}
// A-matrix 16x32 bf16 k order used here (contiguous, kk = 2v + 16*half): VGPR v holds K = kk, kk+1
__device__ __forceinline__ int a_kk(int v, int half) {
    return 2 * v + 16 * half;
}
// B-matrix 32x16 bf16 k order used here (contiguous, same as a_kk); dwords are
// contiguous in K for fixed half -> compiler merges the 8 loads into 2x b128.
__device__ __forceinline__ int b_kk(int v, int half) {
    return 2 * v + 16 * half;
}
__device__ __forceinline__ v8f wmma_bf16(Frag16 a, Frag16 b, v8f c) {
    return __builtin_amdgcn_wmma_f32_16x16x32_bf16(
        false, a.v, false, b.v, (short)0, c, false, false);
}

// ---------------- sizes ----------------
#define NPAIRS 500000
#define NMOLS  10000
#define INW    80      // 2*A + B
#define F320   320
#define H256   256
#define H128   128
#define BN_EPS 1e-5f

// ---------------- weight transpose+convert: dst[o*kpad+k] = bf16(src[k*dout+o]) ----
__global__ void transpose_bf16_kernel(const float* __restrict__ src,
                                      unsigned short* __restrict__ dst,
                                      int din, int dout, int kpad) {
    int i = blockIdx.x * blockDim.x + threadIdx.x;
    if (i >= dout * kpad) return;
    int o = i / kpad, k = i % kpad;
    float v = (k < din) ? src[k * dout + o] : 0.0f;
    dst[o * kpad + k] = f2bf(v);
}

// ---------------- exclusive scan of segment lengths (single block) ----------------
__global__ __launch_bounds__(1024) void scan_kernel(const int* __restrict__ len,
                                                    int* __restrict__ start, int n) {
    __shared__ int buf[1024];
    __shared__ int carry_s;
    int tid = threadIdx.x;
    if (tid == 0) carry_s = 0;
    __syncthreads();
    for (int base = 0; base < n; base += 1024) {
        int i = base + tid;
        int v = (i < n) ? len[i] : 0;
        buf[tid] = v;
        __syncthreads();
        for (int off = 1; off < 1024; off <<= 1) {
            int t = (tid >= off) ? buf[tid - off] : 0;
            __syncthreads();
            buf[tid] += t;
            __syncthreads();
        }
        int carry = carry_s;
        if (i < n) start[i] = carry + buf[tid] - v;   // exclusive
        if (i == n - 1) start[n] = carry + buf[tid];
        __syncthreads();
        if (tid == 1023) carry_s = carry + buf[1023];
        __syncthreads();
    }
}

// ---------------- front: atom1|atom2|bond -> relu -> bf16 f-tile [16][320] ----------
__device__ __forceinline__ void front_tile(const float* __restrict__ xs,
                                           unsigned short* __restrict__ fsh,
                                           const unsigned short* __restrict__ waT,
                                           const float* __restrict__ atom_b,
                                           const unsigned short* __restrict__ wbT,
                                           const float* __restrict__ bond_b,
                                           int nlo, int half) {
    Frag16 a1, a2, ab;
#pragma unroll
    for (int v = 0; v < 8; ++v) {
        int kk = a_kk(v, half);
        a1.u[v] = pack2(xs[nlo * INW + kk],       xs[nlo * INW + kk + 1]);
        a2.u[v] = pack2(xs[nlo * INW + 32 + kk],  xs[nlo * INW + 32 + kk + 1]);
        ab.u[v] = (kk < 16) ? pack2(xs[nlo * INW + 64 + kk], xs[nlo * INW + 64 + kk + 1]) : 0u;
    }
#pragma unroll
    for (int nt = 0; nt < 8; ++nt) {
        Frag16 b;
#pragma unroll
        for (int v = 0; v < 8; ++v)
            b.u[v] = *(const unsigned int*)&waT[(nt * 16 + nlo) * 32 + b_kk(v, half)];
        v8f c1 = {}; c1 = wmma_bf16(a1, b, c1);
        v8f c2 = {}; c2 = wmma_bf16(a2, b, c2);
        float bias = atom_b[nt * 16 + nlo];
#pragma unroll
        for (int v = 0; v < 8; ++v) {
            float t1 = fmaxf(c1[v] + bias, 0.0f);
            float t2 = fmaxf(c2[v] + bias, 0.0f);
            fsh[(v + 8 * half) * F320 + nt * 16 + nlo]       = f2bf(t1);
            fsh[(v + 8 * half) * F320 + 128 + nt * 16 + nlo] = f2bf(t2);
        }
    }
#pragma unroll
    for (int nt = 0; nt < 4; ++nt) {
        Frag16 b;
#pragma unroll
        for (int v = 0; v < 8; ++v)
            b.u[v] = *(const unsigned int*)&wbT[(nt * 16 + nlo) * 32 + b_kk(v, half)];
        v8f c = {}; c = wmma_bf16(ab, b, c);
        float bias = bond_b[nt * 16 + nlo];
#pragma unroll
        for (int v = 0; v < 8; ++v) {
            float t = fmaxf(c[v] + bias, 0.0f);
            fsh[(v + 8 * half) * F320 + 256 + nt * 16 + nlo] = f2bf(t);
        }
    }
}

__device__ __forceinline__ void build_af10(const unsigned short* __restrict__ fsh,
                                           int nlo, int half, Frag16 (&af)[10]) {
#pragma unroll
    for (int kt = 0; kt < 10; ++kt)
#pragma unroll
        for (int v = 0; v < 8; ++v)
            af[kt].u[v] = *(const unsigned int*)&fsh[nlo * F320 + kt * 32 + a_kk(v, half)];
}

// ---------------- pass 1: BN statistics of y = f @ p1_W + p1_b --------------------
__global__ __launch_bounds__(32) void pass1_kernel(
    const float* __restrict__ x,
    const unsigned short* __restrict__ waT, const float* __restrict__ atom_b,
    const unsigned short* __restrict__ wbT, const float* __restrict__ bond_b,
    const unsigned short* __restrict__ w1T, const float* __restrict__ p1_b,
    float* __restrict__ gsum, float* __restrict__ gsq) {
    __shared__ __align__(16) float xs[16 * INW];
    __shared__ __align__(16) unsigned short fsh[16 * F320];
    const int lane = threadIdx.x;
    const int row0 = blockIdx.x * 16;
    for (int i = lane; i < 16 * INW; i += 32) xs[i] = x[row0 * INW + i];
    __syncthreads();
    const int nlo = lane & 15, half = lane >> 4;
    front_tile(xs, fsh, waT, atom_b, wbT, bond_b, nlo, half);
    __syncthreads();
    Frag16 af[10];
    build_af10(fsh, nlo, half, af);
    for (int nt = 0; nt < 16; ++nt) {
        v8f acc = {};
#pragma unroll
        for (int kt = 0; kt < 10; ++kt) {
            Frag16 b;
#pragma unroll
            for (int v = 0; v < 8; ++v)
                b.u[v] = *(const unsigned int*)&w1T[(nt * 16 + nlo) * F320 + kt * 32 + b_kk(v, half)];
            acc = wmma_bf16(af[kt], b, acc);
        }
        float bias = p1_b[nt * 16 + nlo];
        float s = 0.0f, q = 0.0f;
#pragma unroll
        for (int v = 0; v < 8; ++v) { float t = acc[v] + bias; s += t; q += t * t; }
        s += __shfl_xor(s, 16, 32);
        q += __shfl_xor(q, 16, 32);
        if (lane < 16) {
            atomicAdd(&gsum[nt * 16 + lane], s);
            atomicAdd(&gsq[nt * 16 + lane], q);
        }
    }
}

// ---------------- finalize BN stats -------------------------------------------------
__global__ void finalize_kernel(const float* __restrict__ s, const float* __restrict__ q,
                                float* __restrict__ mean, float* __restrict__ rstd,
                                float invN, int n) {
    int i = blockIdx.x * blockDim.x + threadIdx.x;
    if (i < n) {
        float mu = s[i] * invN;
        float var = q[i] * invN - mu * mu;
        mean[i] = mu;
        rstd[i] = rsqrtf(var + BN_EPS);
    }
}

// ---------------- pass 2: recompute y, BN+relu, @p3, segment mean -------------------
__global__ __launch_bounds__(32) void pass2_kernel(
    const float* __restrict__ x, const int* __restrict__ seg_start,
    const unsigned short* __restrict__ waT, const float* __restrict__ atom_b,
    const unsigned short* __restrict__ wbT, const float* __restrict__ bond_b,
    const unsigned short* __restrict__ w1T, const float* __restrict__ p1_b,
    const float* __restrict__ mean, const float* __restrict__ rstd,
    const float* __restrict__ bn_g, const float* __restrict__ bn_b,
    const unsigned short* __restrict__ w3T, const float* __restrict__ p3_b,
    float* __restrict__ h_pair, int n_rows) {
    __shared__ __align__(16) float xs[16 * INW];
    __shared__ __align__(16) unsigned short fsh[16 * F320];
    __shared__ __align__(16) unsigned short ash[16 * H256];
    const int lane = threadIdx.x;
    const int mol = blockIdx.x;
    const int s0 = seg_start[mol], e0 = seg_start[mol + 1];
    const int len = e0 - s0;
    const int nlo = lane & 15, half = lane >> 4;
    float msum[8];
#pragma unroll
    for (int nt = 0; nt < 8; ++nt) msum[nt] = 0.0f;
    const int ntiles = (len + 15) >> 4;
    for (int t = 0; t < ntiles; ++t) {
        const int r0 = s0 + t * 16;
        for (int i = lane; i < 16 * INW; i += 32) {
            int r = r0 + i / INW;
            int idx = (r < n_rows) ? (r * INW + i % INW) : (i % INW);  // pad rows masked later
            xs[i] = x[idx];
        }
        __syncthreads();
        front_tile(xs, fsh, waT, atom_b, wbT, bond_b, nlo, half);
        __syncthreads();
        Frag16 af[10];
        build_af10(fsh, nlo, half, af);
        // y tiles -> BN -> relu -> bf16 activation tile
        for (int nt = 0; nt < 16; ++nt) {
            v8f acc = {};
#pragma unroll
            for (int kt = 0; kt < 10; ++kt) {
                Frag16 b;
#pragma unroll
                for (int v = 0; v < 8; ++v)
                    b.u[v] = *(const unsigned int*)&w1T[(nt * 16 + nlo) * F320 + kt * 32 + b_kk(v, half)];
                acc = wmma_bf16(af[kt], b, acc);
            }
            int c = nt * 16 + nlo;
            float bias = p1_b[c], mu = mean[c], rs = rstd[c], g = bn_g[c], bb = bn_b[c];
#pragma unroll
            for (int v = 0; v < 8; ++v) {
                float yv = acc[v] + bias;
                float tv = fmaxf((yv - mu) * rs * g + bb, 0.0f);
                ash[(v + 8 * half) * H256 + c] = f2bf(tv);
            }
        }
        __syncthreads();
        // x = a @ p3_W + p3_b, masked segment accumulate
        Frag16 a2f[8];
#pragma unroll
        for (int kt = 0; kt < 8; ++kt)
#pragma unroll
            for (int v = 0; v < 8; ++v)
                a2f[kt].u[v] = *(const unsigned int*)&ash[nlo * H256 + kt * 32 + a_kk(v, half)];
#pragma unroll
        for (int nt = 0; nt < 8; ++nt) {
            v8f acc = {};
#pragma unroll
            for (int kt = 0; kt < 8; ++kt) {
                Frag16 b;
#pragma unroll
                for (int v = 0; v < 8; ++v)
                    b.u[v] = *(const unsigned int*)&w3T[(nt * 16 + nlo) * H256 + kt * 32 + b_kk(v, half)];
                acc = wmma_bf16(a2f[kt], b, acc);
            }
            float bias = p3_b[nt * 16 + nlo];
#pragma unroll
            for (int v = 0; v < 8; ++v) {
                int j = t * 16 + v + 8 * half;
                if (j < len) msum[nt] += acc[v] + bias;
            }
        }
        __syncthreads();
    }
    float invlen = 1.0f / (float)len;
#pragma unroll
    for (int nt = 0; nt < 8; ++nt) {
        float sred = msum[nt] + __shfl_xor(msum[nt], 16, 32);
        if (lane < 16) h_pair[mol * H128 + nt * 16 + lane] = sred * invlen;
    }
}

// ---------------- head GEMM: z = h_pair @ fc1 + b, collect stats --------------------
__global__ __launch_bounds__(32) void head_kernel(
    const float* __restrict__ h_pair,
    const unsigned short* __restrict__ wfc1T, const float* __restrict__ fc1_b,
    unsigned short* __restrict__ zbuf,
    float* __restrict__ hsum, float* __restrict__ hsq) {
    __shared__ __align__(16) float hs[16 * H128];
    const int lane = threadIdx.x;
    const int row0 = blockIdx.x * 16;
    for (int i = lane; i < 16 * H128; i += 32) hs[i] = h_pair[row0 * H128 + i];
    __syncthreads();
    const int nlo = lane & 15, half = lane >> 4;
    Frag16 af[4];
#pragma unroll
    for (int kt = 0; kt < 4; ++kt)
#pragma unroll
        for (int v = 0; v < 8; ++v) {
            int kk = kt * 32 + a_kk(v, half);
            af[kt].u[v] = pack2(hs[nlo * H128 + kk], hs[nlo * H128 + kk + 1]);
        }
#pragma unroll
    for (int nt = 0; nt < 8; ++nt) {
        v8f acc = {};
#pragma unroll
        for (int kt = 0; kt < 4; ++kt) {
            Frag16 b;
#pragma unroll
            for (int v = 0; v < 8; ++v)
                b.u[v] = *(const unsigned int*)&wfc1T[(nt * 16 + nlo) * H128 + kt * 32 + b_kk(v, half)];
            acc = wmma_bf16(af[kt], b, acc);
        }
        float bias = fc1_b[nt * 16 + nlo];
        float s = 0.0f, q = 0.0f;
#pragma unroll
        for (int v = 0; v < 8; ++v) {
            float z = acc[v] + bias;
            zbuf[(row0 + v + 8 * half) * H128 + nt * 16 + nlo] = f2bf(z);
            s += z; q += z * z;
        }
        s += __shfl_xor(s, 16, 32);
        q += __shfl_xor(q, 16, 32);
        if (lane < 16) {
            atomicAdd(&hsum[nt * 16 + lane], s);
            atomicAdd(&hsq[nt * 16 + lane], q);
        }
    }
}

// ---------------- final: out = relu(BN(z)) @ fc2 + b --------------------------------
__global__ void out_kernel(const unsigned short* __restrict__ zbuf,
                           const float* __restrict__ mean2, const float* __restrict__ rstd2,
                           const float* __restrict__ g, const float* __restrict__ b,
                           const float* __restrict__ fc2_W, const float* __restrict__ fc2_b,
                           float* __restrict__ out, int m) {
    int r = blockIdx.x * blockDim.x + threadIdx.x;
    if (r >= m) return;
    float acc = fc2_b[0];
    for (int n = 0; n < H128; ++n) {
        float z = bf2f(zbuf[r * H128 + n]);
        float t = fmaxf((z - mean2[n]) * rstd2[n] * g[n] + b[n], 0.0f);
        acc += t * fc2_W[n];
    }
    out[r] = acc;
}

// ---------------- host-side orchestration -------------------------------------------
extern "C" void kernel_launch(void* const* d_in, const int* in_sizes, int n_in,
                              void* d_out, int out_size, void* d_ws, size_t ws_size,
                              hipStream_t stream) {
    (void)in_sizes; (void)n_in; (void)out_size; (void)ws_size;
    const float* x_pair  = (const float*)d_in[0];
    const int*   idx     = (const int*)  d_in[1];
    const float* atom_W  = (const float*)d_in[2];
    const float* atom_b  = (const float*)d_in[3];
    const float* bond_W  = (const float*)d_in[4];
    const float* bond_b  = (const float*)d_in[5];
    const float* p1_W    = (const float*)d_in[6];
    const float* p1_b    = (const float*)d_in[7];
    const float* p_bn_g  = (const float*)d_in[8];
    const float* p_bn_b  = (const float*)d_in[9];
    const float* p3_W    = (const float*)d_in[10];
    const float* p3_b    = (const float*)d_in[11];
    const float* fc1_W   = (const float*)d_in[12];
    const float* fc1_b   = (const float*)d_in[13];
    const float* bn1_g   = (const float*)d_in[14];
    const float* bn1_b   = (const float*)d_in[15];
    const float* fc2_W   = (const float*)d_in[16];
    const float* fc2_b   = (const float*)d_in[17];

    char* ws = (char*)d_ws;
    size_t off = 0;
    auto carve = [&](size_t bytes) -> char* {
        char* p = ws + off;
        off = (off + bytes + 127) & ~((size_t)127);
        return p;
    };
    unsigned short* waT   = (unsigned short*)carve(128 * 32  * 2);
    unsigned short* wbT   = (unsigned short*)carve(64  * 32  * 2);
    unsigned short* w1T   = (unsigned short*)carve(256 * 320 * 2);
    unsigned short* w3T   = (unsigned short*)carve(128 * 256 * 2);
    unsigned short* wfc1T = (unsigned short*)carve(128 * 128 * 2);
    float* gsum  = (float*)carve(256 * 4);
    float* gsq   = (float*)carve(256 * 4);
    float* mean  = (float*)carve(256 * 4);
    float* rstd  = (float*)carve(256 * 4);
    float* hsum  = (float*)carve(128 * 4);
    float* hsq   = (float*)carve(128 * 4);
    float* mean2 = (float*)carve(128 * 4);
    float* rstd2 = (float*)carve(128 * 4);
    int*   seg   = (int*)  carve((NMOLS + 1) * 4);
    float* h_pair = (float*)carve((size_t)NMOLS * H128 * 4);
    unsigned short* zbuf = (unsigned short*)carve((size_t)NMOLS * H128 * 2);

    // zero the accumulation buffers every call (deterministic)
    hipMemsetAsync(gsum, 0, 256 * 4, stream);
    hipMemsetAsync(gsq,  0, 256 * 4, stream);
    hipMemsetAsync(hsum, 0, 128 * 4, stream);
    hipMemsetAsync(hsq,  0, 128 * 4, stream);

    // weight convert/transpose to bf16 [dout][Kpad]
    transpose_bf16_kernel<<<(128 * 32  + 255) / 256, 256, 0, stream>>>(atom_W, waT,   32, 128, 32);
    transpose_bf16_kernel<<<(64  * 32  + 255) / 256, 256, 0, stream>>>(bond_W, wbT,   16, 64,  32);
    transpose_bf16_kernel<<<(256 * 320 + 255) / 256, 256, 0, stream>>>(p1_W,   w1T,  320, 256, 320);
    transpose_bf16_kernel<<<(128 * 256 + 255) / 256, 256, 0, stream>>>(p3_W,   w3T,  256, 128, 256);
    transpose_bf16_kernel<<<(128 * 128 + 255) / 256, 256, 0, stream>>>(fc1_W,  wfc1T, 128, 128, 128);

    scan_kernel<<<1, 1024, 0, stream>>>(idx, seg, NMOLS);

    pass1_kernel<<<NPAIRS / 16, 32, 0, stream>>>(
        x_pair, waT, atom_b, wbT, bond_b, w1T, p1_b, gsum, gsq);

    finalize_kernel<<<1, 256, 0, stream>>>(gsum, gsq, mean, rstd, 1.0f / (float)NPAIRS, 256);

    pass2_kernel<<<NMOLS, 32, 0, stream>>>(
        x_pair, seg, waT, atom_b, wbT, bond_b, w1T, p1_b,
        mean, rstd, p_bn_g, p_bn_b, w3T, p3_b, h_pair, NPAIRS);

    head_kernel<<<NMOLS / 16, 32, 0, stream>>>(h_pair, wfc1T, fc1_b, zbuf, hsum, hsq);

    finalize_kernel<<<1, 128, 0, stream>>>(hsum, hsq, mean2, rstd2, 1.0f / (float)NMOLS, 128);

    out_kernel<<<(NMOLS + 255) / 256, 256, 0, stream>>>(
        zbuf, mean2, rstd2, bn1_g, bn1_b, fc2_W, fc2_b, (float*)d_out, NMOLS);
}
